// GCN_GMP_14903536517934
// MI455X (gfx1250) — matrix operand order repaired and checked by reading, not compiled, not run on hardware
//
#include <hip/hip_runtime.h>
#include <math.h>
#include <stdint.h>

// Problem constants (match reference)
#define NB      32          // graphs
#define NN      300         // nodes per graph
#define FIN     16928
#define HD      2048
#define H2D     1024
#define COUT    10
#define BN_EPS  0.2f
#define RAD2    16.0f       // R^2

typedef __attribute__((ext_vector_type(16))) __bf16 v16bf;
typedef __attribute__((ext_vector_type(8)))  __bf16 v8bf;
typedef __attribute__((ext_vector_type(8)))  float  v8f;
typedef __attribute__((ext_vector_type(4)))  float  f4;
typedef unsigned int u32;
typedef __attribute__((ext_vector_type(4)))  u32 u32x4;

// Async global->LDS staging via CDNA5 inline asm (portable across toolchains,
// avoids guessing clang builtin signatures). Tracked by ASYNCcnt.
#if defined(__gfx1250__)
#define USE_ASYNC 1
#define WAIT_ASYNC(N) asm volatile("s_wait_asynccnt " #N ::: "memory")
#else
#define USE_ASYNC 0
#define WAIT_ASYNC(N)
#endif

union V16U { v16bf v; v8bf h[2]; };

// 16-byte global -> LDS copy.
// vdst VGPR = wave-relative LDS byte offset (= low 32 bits of the flat
// pointer to a __shared__ object, per ISA aperture rules §10.2).
static __device__ __forceinline__ void cp16(const __bf16* g, __bf16* l) {
#if USE_ASYNC
  const unsigned lds = (unsigned)(uintptr_t)l;
  asm volatile("global_load_async_to_lds_b128 %0, %1, off"
               :: "v"(lds), "v"((unsigned long long)(uintptr_t)g)
               : "memory");
#else
  *(u32x4*)l = *(const u32x4*)g;
#endif
}

// -------------------------------------------------------------------------
// Streaming fp32 -> bf16 convert (one-time, HBM-bound)
// -------------------------------------------------------------------------
__global__ __launch_bounds__(256)
void cvt_bf16_kernel(const float* __restrict__ in, __bf16* __restrict__ out, long n)
{
  long i0 = ((long)blockIdx.x * 256 + threadIdx.x) * 8;
  const long stride = (long)gridDim.x * 256 * 8;
  for (long i = i0; i < n; i += stride) {
    f4 a = *(const f4*)&in[i];
    f4 b = *(const f4*)&in[i + 4];
    v8bf o;
    o[0] = (__bf16)a[0]; o[1] = (__bf16)a[1]; o[2] = (__bf16)a[2]; o[3] = (__bf16)a[3];
    o[4] = (__bf16)b[0]; o[5] = (__bf16)b[1]; o[6] = (__bf16)b[2]; o[7] = (__bf16)b[3];
    *(v8bf*)&out[i] = o;
  }
}

// -------------------------------------------------------------------------
// Transpose + convert: in[K][N] fp32  ->  out[N][K] bf16 (LDS 32x32 tiles)
// -------------------------------------------------------------------------
__global__ __launch_bounds__(256)
void transpose_cvt_kernel(const float* __restrict__ in, __bf16* __restrict__ out,
                          int K, int N)
{
  __shared__ float tile[32][33];
  const int n0 = blockIdx.x * 32, k0 = blockIdx.y * 32;
  const int tx = threadIdx.x & 31, ty = threadIdx.x >> 5;   // ty: 0..7
  #pragma unroll
  for (int i = 0; i < 32; i += 8)
    tile[ty + i][tx] = in[(size_t)(k0 + ty + i) * N + n0 + tx];
  __syncthreads();
  #pragma unroll
  for (int i = 0; i < 32; i += 8)
    out[(size_t)(n0 + ty + i) * K + k0 + tx] = (__bf16)tile[tx][ty + i];
}

// -------------------------------------------------------------------------
// GEMM: C[M,N] = epi(A[M,K] @ B[K,N]), A bf16 row-major, BT = B^T bf16
// row-major ([N][K]). Block tile 128x128, BK=32, 256 threads = 8 wave32
// (2 M x 4 N); each wave owns 64x32 = 4x2 WMMA fragments.
// LDS double-buffered; tiles streamed with global_load_async_to_lds_b128
// (4 x 512B wave-loads per tile per wave), retired via s_wait_asynccnt.
// -------------------------------------------------------------------------
template<bool RELU_BN, typename OutT>
__global__ __launch_bounds__(256)
void gemm_bf16_wmma(const __bf16* __restrict__ A, const __bf16* __restrict__ BT,
                    OutT* __restrict__ C, int M, int Nn, int K,
                    const float* __restrict__ gam, const float* __restrict__ bet,
                    const float* __restrict__ rmean, const float* __restrict__ rvar)
{
  constexpr int LDP = 40;                    // padded row stride (halves) = 80B
  __shared__ __bf16 Ald[2][128 * LDP];
  __shared__ __bf16 Bld[2][128 * LDP];

  const int tid  = threadIdx.x;
  const int lane = tid & 31;
  const int wave = tid >> 5;
  const int wm   = wave >> 2;                // 0..1 -> M offset 64*wm
  const int wn   = wave & 3;                 // 0..3 -> N offset 32*wn
  const int m0   = blockIdx.y * 128;
  const int n0   = blockIdx.x * 128;
  const int l15  = lane & 15;
  const int lhi  = lane >> 4;

  v8f acc[4][2];
  #pragma unroll
  for (int mt = 0; mt < 4; ++mt)
    #pragma unroll
    for (int nt = 0; nt < 2; ++nt)
      #pragma unroll
      for (int v = 0; v < 8; ++v) acc[mt][nt][v] = 0.f;

  // Staging map: tile = 128 rows x 64B. 16 segments of (8 rows x 64B = 512B);
  // 2 segments per wave per matrix -> 4 async b128 per wave per K-tile.
  const int srow0 = wave * 16 + (lane >> 2);
  const int srow1 = srow0 + 8;
  const int skh   = (lane & 3) * 8;          // k offset in halves (16B units)
  const __bf16* gA0 = A  + (size_t)(m0 + srow0) * K + skh;
  const __bf16* gA1 = A  + (size_t)(m0 + srow1) * K + skh;
  const __bf16* gB0 = BT + (size_t)(n0 + srow0) * K + skh;
  const __bf16* gB1 = BT + (size_t)(n0 + srow1) * K + skh;
  const int la0 = srow0 * LDP + skh;
  const int la1 = srow1 * LDP + skh;

  auto issue = [&](int kt, int buf) {
    const size_t ko = (size_t)kt * 32;
    cp16(gA0 + ko, &Ald[buf][la0]);
    cp16(gA1 + ko, &Ald[buf][la1]);
    cp16(gB0 + ko, &Bld[buf][la0]);
    cp16(gB1 + ko, &Bld[buf][la1]);
  };

  const int nk = K / 32;
  issue(0, 0);
  if (nk > 1) issue(1, 1);

  for (int kt = 0; kt < nk; ++kt) {
    if (kt == nk - 1) { WAIT_ASYNC(0); }     // no younger tile in flight
    else             { WAIT_ASYNC(4); }      // tile kt done; kt+1 may stream
    __syncthreads();

    const __bf16* Ab = Ald[kt & 1];
    const __bf16* Bb = Bld[kt & 1];

    // Fragments per ISA 7.12.2 (A 16x32 bf16; B 32x16 bf16, col-contiguous K)
    V16U af[4], bfr[2];
    #pragma unroll
    for (int mt = 0; mt < 4; ++mt) {
      const int row = wm * 64 + mt * 16 + l15;
      af[mt].h[0] = *(const v8bf*)&Ab[row * LDP + lhi * 8];
      af[mt].h[1] = *(const v8bf*)&Ab[row * LDP + 16 + lhi * 8];
    }
    #pragma unroll
    for (int nt = 0; nt < 2; ++nt) {
      const int col = wn * 32 + nt * 16 + l15;
      bfr[nt].h[0] = *(const v8bf*)&Bb[col * LDP + lhi * 8];
      bfr[nt].h[1] = *(const v8bf*)&Bb[col * LDP + 16 + lhi * 8];
    }
    #pragma unroll
    for (int mt = 0; mt < 4; ++mt)
      #pragma unroll
      for (int nt = 0; nt < 2; ++nt)
        acc[mt][nt] = __builtin_amdgcn_wmma_f32_16x16x32_bf16(
            false, af[mt].v, false, bfr[nt].v, (short)0, acc[mt][nt], false, false);

    __syncthreads();                         // all waves done reading this buf
    if (kt + 2 < nk) issue(kt + 2, kt & 1);  // refill freed buffer
  }

  // Epilogue: optional ReLU -> BatchNorm(eval); store fp32 or bf16.
  #pragma unroll
  for (int nt = 0; nt < 2; ++nt) {
    const int col = n0 + wn * 32 + nt * 16 + l15;
    float sc = 1.f, sh = 0.f;
    if (RELU_BN) {
      sc = gam[col] * rsqrtf(rvar[col] + BN_EPS);
      sh = bet[col] - rmean[col] * sc;
    }
    #pragma unroll
    for (int mt = 0; mt < 4; ++mt) {
      const int rbase = m0 + wm * 64 + mt * 16 + lhi * 8;
      #pragma unroll
      for (int v = 0; v < 8; ++v) {
        float val = acc[mt][nt][v];
        if (RELU_BN) val = fmaxf(val, 0.f) * sc + sh;
        C[(size_t)(rbase + v) * Nn + col] = (OutT)val;
      }
    }
  }
}

// -------------------------------------------------------------------------
// Degree (with self loop) -> dinv = rsqrt(deg). One block per graph.
// -------------------------------------------------------------------------
__global__ __launch_bounds__(320)
void deg_kernel(const float* __restrict__ pos, float* __restrict__ dinv)
{
  __shared__ float px[NN], py[NN];
  const int b = blockIdx.x, t = threadIdx.x;
  for (int i = t; i < NN; i += 320) {
    px[i] = pos[(size_t)(b * NN + i) * 2 + 0];
    py[i] = pos[(size_t)(b * NN + i) * 2 + 1];
  }
  __syncthreads();
  if (t < NN) {
    const float xi = px[t], yi = py[t];
    int cnt = 1;
    for (int j = 0; j < NN; ++j) {
      const float dx = px[j] - xi, dy = py[j] - yi;
      if (j != t && dx * dx + dy * dy <= RAD2) ++cnt;
    }
    dinv[b * NN + t] = rsqrtf((float)cnt);
  }
}

// -------------------------------------------------------------------------
// a_norm[b][i][j] = (edge || i==j) ? dinv_i * dinv_j : 0
// -------------------------------------------------------------------------
__global__ __launch_bounds__(128)
void anorm_kernel(const float* __restrict__ pos, const float* __restrict__ dinv,
                  float* __restrict__ an)
{
  const int b = blockIdx.y, i = blockIdx.x, t = threadIdx.x;
  const float xi = pos[(size_t)(b * NN + i) * 2 + 0];
  const float yi = pos[(size_t)(b * NN + i) * 2 + 1];
  const float di = dinv[b * NN + i];
  for (int j = t; j < NN; j += 128) {
    const float dx = pos[(size_t)(b * NN + j) * 2 + 0] - xi;
    const float dy = pos[(size_t)(b * NN + j) * 2 + 1] - yi;
    const bool e = (dx * dx + dy * dy <= RAD2) || (j == i);
    an[((size_t)b * NN + i) * NN + j] = e ? di * dinv[b * NN + j] : 0.f;
  }
}

// -------------------------------------------------------------------------
// Fused: m = relu(a_norm @ hW + bc); /sqrt(N); mean-pool. hW tile (300x128
// fp32 = 153.6 KB) lives in dynamic LDS (MI455X: 320 KB/WGP), reused 300x.
// -------------------------------------------------------------------------
__global__ __launch_bounds__(128)
void spmm_pool_kernel(const float* __restrict__ hW, const float* __restrict__ an,
                      const float* __restrict__ bc, float* __restrict__ pooled)
{
  extern __shared__ float sh[];            // [300][128]
  __shared__ float arow[NN];
  const int b = blockIdx.y, f0 = blockIdx.x * 128, t = threadIdx.x;

  for (int idx = t; idx < NN * 128; idx += 128) {
    const int j = idx >> 7, f = idx & 127;
    sh[idx] = hW[(size_t)(b * NN + j) * HD + f0 + f];
  }
  const float bias = bc[f0 + t];
  const float* anb = an + (size_t)b * NN * NN;
  float acc = 0.f;
  for (int i = 0; i < NN; ++i) {
    __syncthreads();
    for (int j = t; j < NN; j += 128) arow[j] = anb[(size_t)i * NN + j];
    __syncthreads();
    float s = bias;
    #pragma unroll 4
    for (int j = 0; j < NN; ++j) s += arow[j] * sh[j * 128 + t];
    acc += fmaxf(s, 0.f);
  }
  pooled[(size_t)b * HD + f0 + t] = acc * (1.f / (300.f * 17.320508075688772f));
}

// -------------------------------------------------------------------------
__global__ __launch_bounds__(256)
void lin2_kernel(const float* __restrict__ pooled, const float* __restrict__ W2,
                 const float* __restrict__ g2, const float* __restrict__ b2,
                 const float* __restrict__ rm2, const float* __restrict__ rv2,
                 float* __restrict__ z)
{
  const int idx = blockIdx.x * 256 + threadIdx.x;
  const int b = idx >> 10, n = idx & 1023;
  const float* pr = pooled + (size_t)b * HD;
  float acc = 0.f;
  for (int k = 0; k < HD; ++k) acc += pr[k] * W2[(size_t)k * H2D + n];
  const float r  = fmaxf(acc, 0.f);
  const float sc = g2[n] * rsqrtf(rv2[n] + BN_EPS);
  z[idx] = (r - rm2[n]) * sc + b2[n];
}

__global__ __launch_bounds__(320)
void lin3_kernel(const float* __restrict__ z, const float* __restrict__ W3,
                 const float* __restrict__ b3, float* __restrict__ out)
{
  const int t = threadIdx.x;               // 320 = 32*10
  const int b = t / 10, c = t % 10;
  const float* zr = z + (size_t)b * H2D;
  float acc = b3[c];
  for (int k = 0; k < H2D; ++k) acc += zr[k] * W3[k * COUT + c];
  out[t] = acc;
}

// -------------------------------------------------------------------------
extern "C" void kernel_launch(void* const* d_in, const int* in_sizes, int n_in,
                              void* d_out, int out_size, void* d_ws, size_t ws_size,
                              hipStream_t stream)
{
  (void)in_sizes; (void)n_in; (void)out_size; (void)ws_size;
  const float* x   = (const float*)d_in[0];
  const float* pos = (const float*)d_in[1];
  // d_in[2] = batch (int64) unused: uniform graph sizes
  const float* W01 = (const float*)d_in[3];
  const float* g1  = (const float*)d_in[4];
  const float* b1  = (const float*)d_in[5];
  const float* rm1 = (const float*)d_in[6];
  const float* rv1 = (const float*)d_in[7];
  const float* Wc  = (const float*)d_in[8];
  const float* bc  = (const float*)d_in[9];
  const float* W2  = (const float*)d_in[10];
  const float* g2  = (const float*)d_in[11];
  const float* b2  = (const float*)d_in[12];
  const float* rm2 = (const float*)d_in[13];
  const float* rv2 = (const float*)d_in[14];
  const float* W3  = (const float*)d_in[15];
  const float* b3  = (const float*)d_in[16];
  float* out = (float*)d_out;

  // Workspace carve-up (~533 MB)
  char* p = (char*)d_ws;
  __bf16* xb    = (__bf16*)p;  p += (size_t)NB * NN * FIN * 2;   // x in bf16
  __bf16* W01T  = (__bf16*)p;  p += (size_t)HD * FIN * 2;        // W01^T bf16
  __bf16* WcT   = (__bf16*)p;  p += (size_t)HD * HD * 2;         // Wc^T bf16
  __bf16* hb    = (__bf16*)p;  p += (size_t)NB * NN * HD * 2;    // h bf16
  float*  hW    = (float*)p;   p += (size_t)NB * NN * HD * 4;    // h@Wc fp32
  float*  an    = (float*)p;   p += (size_t)NB * NN * NN * 4;    // a_norm
  float*  dinv  = (float*)p;   p += (size_t)NB * NN * 4;
  float*  pooled= (float*)p;   p += (size_t)NB * HD * 4;
  float*  z     = (float*)p;

  // One-time precision/layout prep (HBM streaming, negligible vs GEMM1)
  cvt_bf16_kernel<<<8192, 256, 0, stream>>>(x, xb, (long)NB * NN * FIN);
  transpose_cvt_kernel<<<dim3(HD / 32, FIN / 32), 256, 0, stream>>>(W01, W01T, FIN, HD);
  transpose_cvt_kernel<<<dim3(HD / 32, HD / 32), 256, 0, stream>>>(Wc, WcT, HD, HD);

  deg_kernel<<<NB, 320, 0, stream>>>(pos, dinv);
  anorm_kernel<<<dim3(NN, NB), 128, 0, stream>>>(pos, dinv, an);

  // h = BN(relu(x @ W01)) : 9600x16928x2048 bf16 WMMA, bf16 out
  gemm_bf16_wmma<true, __bf16><<<dim3(HD / 128, (NB * NN) / 128), 256, 0, stream>>>(
      xb, W01T, hb, NB * NN, HD, FIN, g1, b1, rm1, rv1);

  // hW = h @ Wc : 9600x2048x2048, fp32 out
  gemm_bf16_wmma<false, float><<<dim3(HD / 128, (NB * NN) / 128), 256, 0, stream>>>(
      hb, WcT, hW, NB * NN, HD, HD, nullptr, nullptr, nullptr, nullptr);

  spmm_pool_kernel<<<dim3(HD / 128, NB), 128, NN * 128 * sizeof(float), stream>>>(
      hW, an, bc, pooled);
  lin2_kernel<<<(NB * H2D) / 256, 256, 0, stream>>>(pooled, W2, g2, b2, rm2, rv2, z);
  lin3_kernel<<<1, 320, 0, stream>>>(z, W3, b3, out);
}
